// GAT_45707041964862
// MI455X (gfx1250) — compile-verified, matrix operand order checked by reading, not run on hardware
//
#include <hip/hip_runtime.h>
#include <hip/hip_bf16.h>
#include <math.h>

// ---------------------------------------------------------------------------
// GAT forward (2 layers) for MI455X / gfx1250.
// GEMMs use v_wmma_f32_16x16x32_bf16 (bf16 A/B, f32 accumulate).
// Edge softmax/aggregation: 3-pass atomics (encoded-uint atomicMax for
// segment-max; f32 atomicAdd for denom + aggregation). All data hot in 192MB L2.
// ---------------------------------------------------------------------------

#define N_NODES 50000
#define N_EDGES 800000
#define E_TOT   (N_EDGES + N_NODES)   // self-loops appended
#define IN_F    256
#define HID     64
#define HEADS   4
#define LABELS  32
#define HC1     (HEADS * HID)     // 256
#define HC2     (HEADS * LABELS)  // 128
#define GAT_EPS 1e-16f

typedef __attribute__((ext_vector_type(16))) __bf16 v16bf;
typedef __attribute__((ext_vector_type(8)))  float  v8f;

union V16U { v16bf v; unsigned short u[16]; };

// ---- helpers ---------------------------------------------------------------

__device__ __forceinline__ unsigned short f32_to_bf16(float f) {
    unsigned u = __float_as_uint(f);
    // round-to-nearest-even truncation to bf16
    unsigned r = u + 0x7FFFu + ((u >> 16) & 1u);
    return (unsigned short)(r >> 16);
}

__device__ __forceinline__ unsigned enc_f32(float f) {
    unsigned u = __float_as_uint(f);
    return (u & 0x80000000u) ? ~u : (u | 0x80000000u);
}
__device__ __forceinline__ float dec_f32(unsigned e) {
    return (e & 0x80000000u) ? __uint_as_float(e ^ 0x80000000u)
                             : __uint_as_float(~e);
}
#define ENC_NEG_INF 0x007FFFFFu   // enc(-inf): everything compares greater

__device__ __forceinline__ void edge_endpoints(int e, const int* ei,
                                               int* s, int* d) {
    if (e < N_EDGES) { *s = ei[e]; *d = ei[N_EDGES + e]; }
    else             { *s = *d = e - N_EDGES; }
}

// ---- conversion kernels ----------------------------------------------------

__global__ void k_cvt_bf16(const float* __restrict__ in,
                           unsigned short* __restrict__ out, int n) {
    int i = blockIdx.x * blockDim.x + threadIdx.x;
    if (i < n) out[i] = f32_to_bf16(in[i]);
}

// W row-major [K][M] f32  ->  Wt [M][K] bf16
__global__ void k_cvt_w_t(const float* __restrict__ W,
                          unsigned short* __restrict__ Wt, int K, int M) {
    int i = blockIdx.x * blockDim.x + threadIdx.x;
    if (i >= K * M) return;
    int k = i / M, m = i % M;
    Wt[m * K + k] = f32_to_bf16(W[k * M + m]);
}

// ---- WMMA GEMM: C[M x Ncols] = A(bf16, row-major MxK) * Bt(bf16, [Ncols][K])
// one wave (32 threads) per 16x16 tile; K stepped by 32
__global__ void k_gemm_bf16_wmma(const unsigned short* __restrict__ A,
                                 const unsigned short* __restrict__ Bt,
                                 float* __restrict__ C,
                                 int K, int Ncols) {
    const int lane = threadIdx.x;        // 0..31
    const int half = lane >> 4;          // 0: K+0..7/16..23, 1: K+8..15/24..31
    const int l    = lane & 15;
    const int arow = blockIdx.x * 16 + l;      // A row held by this lane
    const int bcol = blockIdx.y * 16 + l;      // B column held by this lane

    const unsigned short* ap = A  + (size_t)arow * K + half * 8;
    const unsigned short* bp = Bt + (size_t)bcol * K + half * 8;

    v8f acc = {};
    for (int k0 = 0; k0 < K; k0 += 32) {
        V16U a, b;
#pragma unroll
        for (int i = 0; i < 8; ++i) {
            a.u[i]     = ap[k0 + i];
            a.u[i + 8] = ap[k0 + i + 16];
            b.u[i]     = bp[k0 + i];
            b.u[i + 8] = bp[k0 + i + 16];
        }
        acc = __builtin_amdgcn_wmma_f32_16x16x32_bf16(
                  false, a.v, false, b.v, (short)0, acc, false, false);
    }
    // D layout: VGPR j -> row = tileM*16 + j + 8*half, col = bcol
    const int mrow = blockIdx.x * 16 + half * 8;
#pragma unroll
    for (int j = 0; j < 8; ++j)
        C[(size_t)(mrow + j) * Ncols + bcol] = acc[j];
}

// ---- attention scores: a_s[n,h] = dot(h[n,h,:], att_src[h,:]) --------------

__global__ void k_attn_scores(const float* __restrict__ h,
                              const float* __restrict__ att_src,
                              const float* __restrict__ att_dst,
                              float* __restrict__ a_s,
                              float* __restrict__ a_d,
                              int C, int HCtot) {
    int i = blockIdx.x * blockDim.x + threadIdx.x;
    if (i >= N_NODES * HEADS) return;
    int n = i / HEADS, hd = i % HEADS;
    const float* hp = h + (size_t)n * HCtot + hd * C;
    const float* as = att_src + hd * C;
    const float* ad = att_dst + hd * C;
    float s = 0.f, d = 0.f;
    for (int c = 0; c < C; ++c) { s += hp[c] * as[c]; d += hp[c] * ad[c]; }
    a_s[i] = s; a_d[i] = d;
}

// ---- softmax passes over edges --------------------------------------------

__global__ void k_init_menc(unsigned* __restrict__ m_enc, int n) {
    int i = blockIdx.x * blockDim.x + threadIdx.x;
    if (i < n) m_enc[i] = ENC_NEG_INF;
}

__global__ void k_edge_max(const int* __restrict__ ei,
                           const float* __restrict__ a_s,
                           const float* __restrict__ a_d,
                           unsigned* __restrict__ m_enc) {
    int i = blockIdx.x * blockDim.x + threadIdx.x;
    if (i >= E_TOT * HEADS) return;
    int e = i / HEADS, hd = i % HEADS;
    int s, d; edge_endpoints(e, ei, &s, &d);
    float v = a_s[s * HEADS + hd] + a_d[d * HEADS + hd];
    v = (v > 0.f) ? v : 0.2f * v;                    // leaky_relu(0.2)
    atomicMax(&m_enc[d * HEADS + hd], enc_f32(v));
}

__global__ void k_edge_expsum(const int* __restrict__ ei,
                              const float* __restrict__ a_s,
                              const float* __restrict__ a_d,
                              const unsigned* __restrict__ m_enc,
                              float* __restrict__ denom,
                              float* __restrict__ num) {
    int i = blockIdx.x * blockDim.x + threadIdx.x;
    if (i >= E_TOT * HEADS) return;
    int e = i / HEADS, hd = i % HEADS;
    int s, d; edge_endpoints(e, ei, &s, &d);
    float v = a_s[s * HEADS + hd] + a_d[d * HEADS + hd];
    v = (v > 0.f) ? v : 0.2f * v;
    float m  = dec_f32(m_enc[d * HEADS + hd]);
    float ee = expf(v - m);
    num[i] = ee;
    atomicAdd(&denom[d * HEADS + hd], ee);
}

// out[dst, c] += h[src, c] * num[e,h] / (denom[dst,h] + eps)
__global__ void k_edge_aggregate(const int* __restrict__ ei,
                                 const float* __restrict__ hsrc,
                                 const float* __restrict__ num,
                                 const float* __restrict__ denom,
                                 float* __restrict__ out,
                                 int C, int HCtot) {
    unsigned i = blockIdx.x * blockDim.x + threadIdx.x;
    unsigned total = (unsigned)E_TOT * (unsigned)HCtot;
    if (i >= total) return;
    int e  = i / HCtot;
    int c  = i % HCtot;
    int hd = c / C;
    int s, d; edge_endpoints(e, ei, &s, &d);
    float alpha = num[e * HEADS + hd] / (denom[d * HEADS + hd] + GAT_EPS);
    atomicAdd(&out[(size_t)d * HCtot + c], hsrc[(size_t)s * HCtot + c] * alpha);
}

// ---- epilogues -------------------------------------------------------------

__global__ void k_bias_elu_cvt(const float* __restrict__ agg,
                               const float* __restrict__ b,
                               unsigned short* __restrict__ out_bf16) {
    int i = blockIdx.x * blockDim.x + threadIdx.x;
    if (i >= N_NODES * HC1) return;
    float v = agg[i] + b[i % HC1];
    v = (v > 0.f) ? v : expm1f(v);                   // elu
    out_bf16[i] = f32_to_bf16(v);
}

__global__ void k_bias_sigmoid(float* __restrict__ out,
                               const float* __restrict__ b) {
    int i = blockIdx.x * blockDim.x + threadIdx.x;
    if (i >= N_NODES * HC2) return;
    float v = out[i] + b[i % HC2];
    out[i] = 1.f / (1.f + expf(-v));
}

// ---------------------------------------------------------------------------

extern "C" void kernel_launch(void* const* d_in, const int* in_sizes, int n_in,
                              void* d_out, int out_size, void* d_ws, size_t ws_size,
                              hipStream_t stream) {
    const float* x        = (const float*)d_in[0];
    const int*   ei       = (const int*)  d_in[1];
    const float* W1       = (const float*)d_in[2];
    const float* att_src1 = (const float*)d_in[3];
    const float* att_dst1 = (const float*)d_in[4];
    const float* b1       = (const float*)d_in[5];
    const float* W2       = (const float*)d_in[6];
    const float* att_src2 = (const float*)d_in[7];
    const float* att_dst2 = (const float*)d_in[8];
    const float* b2       = (const float*)d_in[9];
    float* out = (float*)d_out;

    // ---- workspace layout (phase-reused) ----
    char* ws = (char*)d_ws;
    size_t off = 0;
    auto carve = [&](size_t bytes) { size_t o = off; off += (bytes + 255) & ~(size_t)255; return o; };
    float*          h1     = (float*)(ws + carve((size_t)N_NODES * HC1 * 4)); // GEMM1 out; later h2 aliases
    float*          agg1   = (float*)(ws + carve((size_t)N_NODES * HC1 * 4));
    unsigned short* xb     = (unsigned short*)(ws + carve((size_t)N_NODES * IN_F * 2)); // x_bf16, reused as elu(h)_bf16
    float*          num    = (float*)(ws + carve((size_t)E_TOT * HEADS * 4));
    float*          a_s    = (float*)(ws + carve((size_t)N_NODES * HEADS * 4));
    float*          a_d    = (float*)(ws + carve((size_t)N_NODES * HEADS * 4));
    unsigned*       m_enc  = (unsigned*)(ws + carve((size_t)N_NODES * HEADS * 4));
    float*          denom  = (float*)(ws + carve((size_t)N_NODES * HEADS * 4));
    unsigned short* w1t    = (unsigned short*)(ws + carve((size_t)IN_F * HC1 * 2));
    unsigned short* w2t    = (unsigned short*)(ws + carve((size_t)HC1 * HC2 * 2));
    float*          h2     = h1;  // layer-2 GEMM output aliases h1 region
    (void)ws_size; (void)n_in; (void)in_sizes; (void)out_size;

    const int TB = 256;
    auto blocks = [&](long long n) { return (unsigned)((n + TB - 1) / TB); };
    const int NH = N_NODES * HEADS;
    const long long EH = (long long)E_TOT * HEADS;

    // ================= Layer 1 =================
    k_cvt_bf16<<<blocks((long long)N_NODES * IN_F), TB, 0, stream>>>(x, xb, N_NODES * IN_F);
    k_cvt_w_t <<<blocks(IN_F * HC1), TB, 0, stream>>>(W1, w1t, IN_F, HC1);

    {   dim3 g(N_NODES / 16, HC1 / 16);
        k_gemm_bf16_wmma<<<g, 32, 0, stream>>>(xb, w1t, h1, IN_F, HC1); }

    k_attn_scores<<<blocks(NH), TB, 0, stream>>>(h1, att_src1, att_dst1, a_s, a_d, HID, HC1);

    hipMemsetAsync(denom, 0, (size_t)NH * 4, stream);
    hipMemsetAsync(agg1,  0, (size_t)N_NODES * HC1 * 4, stream);
    k_init_menc<<<blocks(NH), TB, 0, stream>>>(m_enc, NH);

    k_edge_max   <<<blocks(EH), TB, 0, stream>>>(ei, a_s, a_d, m_enc);
    k_edge_expsum<<<blocks(EH), TB, 0, stream>>>(ei, a_s, a_d, m_enc, denom, num);
    k_edge_aggregate<<<blocks((long long)E_TOT * HC1), TB, 0, stream>>>(
        ei, h1, num, denom, agg1, HID, HC1);

    k_bias_elu_cvt<<<blocks((long long)N_NODES * HC1), TB, 0, stream>>>(agg1, b1, xb);

    // ================= Layer 2 =================
    k_cvt_w_t<<<blocks(HC1 * HC2), TB, 0, stream>>>(W2, w2t, HC1, HC2);

    {   dim3 g(N_NODES / 16, HC2 / 16);
        k_gemm_bf16_wmma<<<g, 32, 0, stream>>>(xb, w2t, h2, HC1, HC2); }

    k_attn_scores<<<blocks(NH), TB, 0, stream>>>(h2, att_src2, att_dst2, a_s, a_d, LABELS, HC2);

    hipMemsetAsync(denom, 0, (size_t)NH * 4, stream);
    hipMemsetAsync(out,   0, (size_t)N_NODES * HC2 * 4, stream);
    k_init_menc<<<blocks(NH), TB, 0, stream>>>(m_enc, NH);

    k_edge_max   <<<blocks(EH), TB, 0, stream>>>(ei, a_s, a_d, m_enc);
    k_edge_expsum<<<blocks(EH), TB, 0, stream>>>(ei, a_s, a_d, m_enc, denom, num);
    k_edge_aggregate<<<blocks((long long)E_TOT * HC2), TB, 0, stream>>>(
        ei, h2, num, denom, out, LABELS, HC2);

    k_bias_sigmoid<<<blocks((long long)N_NODES * HC2), TB, 0, stream>>>(out, b2);
}
